// KnotGraphNet_21320217658011
// MI455X (gfx1250) — compile-verified, matrix operand order checked by reading, not run on hardware
//
#include <hip/hip_runtime.h>
#include <hip/hip_bf16.h>

#define HID      64
#define HEADS    8
#define DH       8
#define LAYERS   6
#define EDGE_DIM 16
#define SIGN_DIM 8

typedef __attribute__((ext_vector_type(16))) _Float16 v16h;
typedef __attribute__((ext_vector_type(8)))  _Float16 v8h;
typedef __attribute__((ext_vector_type(8)))  float    v8f;

__device__ __forceinline__ void atomicMaxF(float* addr, float val) {
  int old = __float_as_int(*addr);
  while (__int_as_float(old) < val) {
    int assumed = old;
    old = atomicCAS((int*)addr, assumed, __float_as_int(val));
    if (old == assumed) break;
  }
}

// Assemble a 16x32 f16 WMMA fragment (A row or B column) from LDS.
// Per CDNA5 ISA 7.12.2 the 16 elements of lane-group g are two contiguous
// 8-half runs: K = kb + g*8 + [0..7] and K = kb + 16 + g*8 + [0..7].
// base points at the start of the 64-half row/column in LDS.
__device__ __forceinline__ v16h frag_from_lds(const _Float16* base, int kb, int g) {
  const v8h lo = *(const v8h*)(base + kb + g * 8);
  const v8h hi = *(const v8h*)(base + kb + 16 + g * 8);
  v16h r;
#pragma unroll
  for (int t = 0; t < 8; ++t) { r[t] = lo[t]; r[t + 8] = hi[t]; }
  return r;
}

// ---------------------------------------------------------------------------
// x[n,h] = sum_s emb_strand[x_nodes[n,s],h] + emb_pos[s,h]
__global__ void node_embed_kernel(const int* __restrict__ x_nodes,
                                  const float* __restrict__ emb_strand,
                                  const float* __restrict__ emb_pos,
                                  float* __restrict__ x, int N) {
  int i = blockIdx.x * blockDim.x + threadIdx.x;
  if (i >= N * HID) return;
  int n = i / HID, h = i % HID;
  float s = 0.f;
#pragma unroll
  for (int sl = 0; sl < 4; ++sl) {
    int tok = x_nodes[n * 4 + sl];
    s += emb_strand[tok * HID + h] + emb_pos[sl * HID + h];
  }
  x[i] = s;
}

// ---------------------------------------------------------------------------
// ef[e,:] = [emb_strand[lab] | emb_sign[sgn]] @ We + be     (72 -> 16)
__global__ void edge_feat_kernel(const int* __restrict__ edge_label,
                                 const int* __restrict__ edge_sign,
                                 const float* __restrict__ emb_strand,
                                 const float* __restrict__ emb_sign,
                                 const float* __restrict__ We,
                                 const float* __restrict__ be,
                                 float* __restrict__ ef, int E) {
  __shared__ float sWe[(HID + SIGN_DIM) * EDGE_DIM];
  for (int i = threadIdx.x; i < (HID + SIGN_DIM) * EDGE_DIM; i += blockDim.x)
    sWe[i] = We[i];
  __syncthreads();
  int i = blockIdx.x * blockDim.x + threadIdx.x;
  if (i >= E * EDGE_DIM) return;
  int e = i / EDGE_DIM, j = i % EDGE_DIM;
  const float* es = emb_strand + (size_t)edge_label[e] * HID;
  const float* eg = emb_sign + (size_t)edge_sign[e] * SIGN_DIM;
  float s = be[j];
  for (int t = 0; t < HID; ++t)      s += es[t] * sWe[t * EDGE_DIM + j];
#pragma unroll
  for (int t = 0; t < SIGN_DIM; ++t) s += eg[t] * sWe[(HID + t) * EDGE_DIM + j];
  ef[i] = s;
}

// ---------------------------------------------------------------------------
// Fused Q/K/V/skip GEMM: x[N,64] @ {Wq,Wk,Wv,Wskip}[64,64] + bias via WMMA f16.
// Block = 16 rows, 8 wave32 waves; wave w -> matrix w/2, column half (w&1)*32.
// Operands staged through LDS as f16 so fragment loads are aligned ds_load_b128.
__global__ void qkvs_wmma_kernel(const float* __restrict__ x,
                                 const float* __restrict__ Wq, const float* __restrict__ bq,
                                 const float* __restrict__ Wk, const float* __restrict__ bk,
                                 const float* __restrict__ Wv, const float* __restrict__ bv,
                                 const float* __restrict__ Ws, const float* __restrict__ bs,
                                 float* __restrict__ q, float* __restrict__ k,
                                 float* __restrict__ v, float* __restrict__ xout, int N) {
  __shared__ __align__(16) _Float16 sx[16 * HID];          // A tile (row-major)
  __shared__ __align__(16) _Float16 swt[4 * HID * HID];    // W^T per matrix: [m][n*64+k]

  const int tid  = threadIdx.x;
  const int lane = tid & 31;
  const int wave = tid >> 5;
  const int mtx  = wave >> 1;
  const int colh = (wave & 1) * 32;
  const int row0 = blockIdx.x * 16;
  const int m    = lane & 15;   // A row within tile / B column within tile
  const int g    = lane >> 4;   // lane group selects K-half

  // ---- stage A tile: 256 threads x one float4 each (row guard once, zero fill)
  {
    int r = tid >> 4;           // 0..15
    int c = (tid & 15) * 4;     // 0,4,...,60
    float4 f = make_float4(0.f, 0.f, 0.f, 0.f);
    if (row0 + r < N) f = *(const float4*)&x[(size_t)(row0 + r) * HID + c];
    sx[r * HID + c + 0] = (_Float16)f.x;
    sx[r * HID + c + 1] = (_Float16)f.y;
    sx[r * HID + c + 2] = (_Float16)f.z;
    sx[r * HID + c + 3] = (_Float16)f.w;
  }
  // ---- stage W matrices transposed (coalesced reads, f16 in LDS)
#pragma unroll
  for (int mm = 0; mm < 4; ++mm) {
    const float* Wsrc = (mm == 0) ? Wq : (mm == 1) ? Wk : (mm == 2) ? Wv : Ws;
    for (int idx = tid; idx < HID * HID; idx += 256) {
      int kk = idx >> 6, nn = idx & 63;
      swt[mm * HID * HID + nn * HID + kk] = (_Float16)Wsrc[idx];
    }
  }
  __syncthreads();

  const float* B; float* O;
  if      (mtx == 0) { B = bq; O = q;    }
  else if (mtx == 1) { B = bk; O = k;    }
  else if (mtx == 2) { B = bv; O = v;    }
  else               { B = bs; O = xout; }
  const _Float16* wbase = swt + mtx * HID * HID;

  // bias is per-column: hoist the two values this lane needs
  const int c0 = colh + m;
  const float bias0 = B[c0];
  const float bias1 = B[c0 + 16];

  v8f acc0 = {}, acc1 = {};
#pragma unroll
  for (int kb = 0; kb < HID; kb += 32) {
    v16h a  = frag_from_lds(&sx[m * HID], kb, g);
    v16h b0 = frag_from_lds(wbase + (colh + m) * HID, kb, g);
    v16h b1 = frag_from_lds(wbase + (colh + 16 + m) * HID, kb, g);
    acc0 = __builtin_amdgcn_wmma_f32_16x16x32_f16(false, a, false, b0,
                                                  (short)0, acc0, false, false);
    acc1 = __builtin_amdgcn_wmma_f32_16x16x32_f16(false, a, false, b1,
                                                  (short)0, acc1, false, false);
  }

  // C layout: VGPR r -> M = r + 8*g, N = lane%16.
  // Fast path: whole 16-row tile in range -> straight coalesced stores.
  if (row0 + 16 <= N) {
    float* Obase = O + (size_t)(row0 + 8 * g) * HID + c0;
#pragma unroll
    for (int r = 0; r < 8; ++r) {
      Obase[(size_t)r * HID]      = acc0[r] + bias0;
      Obase[(size_t)r * HID + 16] = acc1[r] + bias1;
    }
  } else {
#pragma unroll
    for (int r = 0; r < 8; ++r) {
      int orow = row0 + r + 8 * g;
      if (orow < N) {
        O[(size_t)orow * HID + c0]      = acc0[r] + bias0;
        O[(size_t)orow * HID + c0 + 16] = acc1[r] + bias1;
      }
    }
  }
}

// ---------------------------------------------------------------------------
// alpha[e,h] = scale * sum_d q[dst,h,d] * (k[src,h,d] + (ef[e]@Wedge)[h,d]); segment max
__global__ void edge_alpha_kernel(const int* __restrict__ edge_index,
                                  const float* __restrict__ q,
                                  const float* __restrict__ k,
                                  const float* __restrict__ ef,
                                  const float* __restrict__ Wedge,
                                  float* __restrict__ alpha,
                                  float* __restrict__ mmax, int E) {
  __shared__ float sW[EDGE_DIM * HID];
  for (int i = threadIdx.x; i < EDGE_DIM * HID; i += blockDim.x) sW[i] = Wedge[i];
  __syncthreads();
  int i = blockIdx.x * blockDim.x + threadIdx.x;
  if (i >= E * HEADS) return;
  int e = i / HEADS, h = i % HEADS;
  int src = edge_index[e], dst = edge_index[E + e];
  float efr[EDGE_DIM];
#pragma unroll
  for (int t = 0; t < EDGE_DIM; ++t) efr[t] = ef[(size_t)e * EDGE_DIM + t];
  const float* qd = q + (size_t)dst * HID + h * DH;
  const float* ks = k + (size_t)src * HID + h * DH;
  float s = 0.f;
#pragma unroll
  for (int d = 0; d < DH; ++d) {
    float ev = 0.f;
#pragma unroll
    for (int t = 0; t < EDGE_DIM; ++t) ev += efr[t] * sW[t * HID + h * DH + d];
    s += qd[d] * (ks[d] + ev);
  }
  s *= 0.35355339059327373f;  // 1/sqrt(DH)
  alpha[i] = s;
  atomicMaxF(&mmax[dst * HEADS + h], s);
}

// ---------------------------------------------------------------------------
__global__ void edge_expsum_kernel(const int* __restrict__ edge_index,
                                   const float* __restrict__ mmax,
                                   float* __restrict__ alpha,  // in: logits, out: exp
                                   float* __restrict__ ssum, int E) {
  int i = blockIdx.x * blockDim.x + threadIdx.x;
  if (i >= E * HEADS) return;
  int e = i / HEADS, h = i % HEADS;
  int dst = edge_index[E + e];
  float a = __expf(alpha[i] - mmax[dst * HEADS + h]);
  alpha[i] = a;
  atomicAdd(&ssum[dst * HEADS + h], a);
}

// ---------------------------------------------------------------------------
// xout[dst,c] += (v[src,c] + (ef[e]@Wedge)[c]) * a[e,h]/(ssum[dst,h]+1e-16)
__global__ void edge_message_kernel(const int* __restrict__ edge_index,
                                    const float* __restrict__ v,
                                    const float* __restrict__ ef,
                                    const float* __restrict__ Wedge,
                                    const float* __restrict__ a,
                                    const float* __restrict__ ssum,
                                    float* __restrict__ xout, int E) {
  __shared__ float sW[EDGE_DIM * HID];
  for (int i = threadIdx.x; i < EDGE_DIM * HID; i += blockDim.x) sW[i] = Wedge[i];
  __syncthreads();
  int i = blockIdx.x * blockDim.x + threadIdx.x;
  if (i >= E * HID) return;
  int e = i / HID, c = i % HID;
  int h = c / DH;
  int src = edge_index[e], dst = edge_index[E + e];
  if (e + 64 < E) __builtin_prefetch(ef + (size_t)(e + 64) * EDGE_DIM, 0, 1);
  float ev = 0.f;
#pragma unroll
  for (int t = 0; t < EDGE_DIM; ++t) ev += ef[(size_t)e * EDGE_DIM + t] * sW[t * HID + c];
  float w = a[(size_t)e * HEADS + h] / (ssum[dst * HEADS + h] + 1e-16f);
  atomicAdd(&xout[(size_t)dst * HID + c], (v[(size_t)src * HID + c] + ev) * w);
}

// ---------------------------------------------------------------------------
__global__ void init_ms_kernel(float* __restrict__ mmax, float* __restrict__ ssum, int n) {
  int i = blockIdx.x * blockDim.x + threadIdx.x;
  if (i < n) { mmax[i] = -3.0e38f; ssum[i] = 0.f; }
}

__global__ void zero_kernel(float* __restrict__ p, int n) {
  int i = blockIdx.x * blockDim.x + threadIdx.x;
  if (i < n) p[i] = 0.f;
}

// x = relu(layernorm(xout))
__global__ void ln_relu_kernel(const float* __restrict__ xout,
                               const float* __restrict__ g,
                               const float* __restrict__ b,
                               float* __restrict__ x, int N) {
  int n = blockIdx.x * blockDim.x + threadIdx.x;
  if (n >= N) return;
  const float* o = xout + (size_t)n * HID;
  float s1 = 0.f, s2 = 0.f;
  for (int i = 0; i < HID; ++i) { float t = o[i]; s1 += t; s2 += t * t; }
  float mu  = s1 * (1.0f / HID);
  float var = s2 * (1.0f / HID) - mu * mu;
  float inv = rsqrtf(var + 1e-5f);
  for (int i = 0; i < HID; ++i) {
    float t = g[i] * (o[i] - mu) * inv + b[i];
    x[(size_t)n * HID + i] = t > 0.f ? t : 0.f;
  }
}

// ---------------------------------------------------------------------------
__global__ void policy_kernel(const float* __restrict__ x,
                              const float* __restrict__ Wp,
                              const float* __restrict__ bp,
                              float* __restrict__ out, int N) {
  int n = blockIdx.x * blockDim.x + threadIdx.x;
  if (n >= N) return;
  float s0 = bp[0], s1 = bp[1];
  for (int t = 0; t < HID; ++t) {
    float xv = x[(size_t)n * HID + t];
    s0 += xv * Wp[t * 2];
    s1 += xv * Wp[t * 2 + 1];
  }
  out[n * 2]     = s0;
  out[n * 2 + 1] = s1;
}

__global__ void colsum_kernel(const float* __restrict__ x, float* __restrict__ gsum, int N) {
  int c = threadIdx.x;  // 64 columns
  float s = 0.f;
  for (int n = blockIdx.x; n < N; n += gridDim.x) s += x[(size_t)n * HID + c];
  atomicAdd(&gsum[c], s);
}

__global__ void value_kernel(const float* __restrict__ gsum,
                             const float* __restrict__ Wv1, const float* __restrict__ bv1,
                             const float* __restrict__ Wv2, const float* __restrict__ bv2,
                             float* __restrict__ out, int N) {
  __shared__ float sh[HID];
  int j = threadIdx.x;
  float s = bv1[j];
  float invN = 1.0f / (float)N;
  for (int t = 0; t < HID; ++t) s += (gsum[t] * invN) * Wv1[t * HID + j];
  s = s > 0.f ? s : 0.f;
  sh[j] = s * Wv2[j];
  __syncthreads();
  if (j == 0) {
    float acc = bv2[0];
    for (int t = 0; t < HID; ++t) acc += sh[t];
    out[0] = tanhf(acc);
  }
}

// ---------------------------------------------------------------------------
extern "C" void kernel_launch(void* const* d_in, const int* in_sizes, int n_in,
                              void* d_out, int out_size, void* d_ws, size_t ws_size,
                              hipStream_t stream) {
  const int*   x_nodes    = (const int*)  d_in[0];
  const int*   edge_index = (const int*)  d_in[1];
  const int*   edge_label = (const int*)  d_in[2];
  const int*   edge_sign  = (const int*)  d_in[3];
  const float* emb_strand = (const float*)d_in[4];
  const float* emb_pos    = (const float*)d_in[5];
  const float* emb_sign   = (const float*)d_in[6];
  const float* We         = (const float*)d_in[7];
  const float* be         = (const float*)d_in[8];
  const float* Wq         = (const float*)d_in[9];
  const float* bq         = (const float*)d_in[10];
  const float* Wk         = (const float*)d_in[11];
  const float* bk         = (const float*)d_in[12];
  const float* Wv         = (const float*)d_in[13];
  const float* bv         = (const float*)d_in[14];
  const float* Wedge      = (const float*)d_in[15];
  const float* Wskip      = (const float*)d_in[16];
  const float* bskip      = (const float*)d_in[17];
  const float* ln_g       = (const float*)d_in[18];
  const float* ln_b       = (const float*)d_in[19];
  const float* Wp         = (const float*)d_in[20];
  const float* bp         = (const float*)d_in[21];
  const float* Wv1        = (const float*)d_in[22];
  const float* bv1        = (const float*)d_in[23];
  const float* Wv2        = (const float*)d_in[24];
  const float* bv2        = (const float*)d_in[25];

  const int N = in_sizes[0] / 4;
  const int E = in_sizes[2];

  float* ws = (float*)d_ws;
  size_t off = 0;
  float* xA   = ws + off; off += (size_t)N * HID;       // layer input
  float* xB   = ws + off; off += (size_t)N * HID;       // layer output (skip + messages)
  float* qbuf = ws + off; off += (size_t)N * HID;
  float* kbuf = ws + off; off += (size_t)N * HID;
  float* vbuf = ws + off; off += (size_t)N * HID;
  float* efb  = ws + off; off += (size_t)E * EDGE_DIM;
  float* abuf = ws + off; off += (size_t)E * HEADS;     // logits then exp weights
  float* mbuf = ws + off; off += (size_t)N * HEADS;
  float* sbuf = ws + off; off += (size_t)N * HEADS;
  float* gsum = ws + off; off += HID;

  const int T = 256;
  node_embed_kernel<<<(N * HID + T - 1) / T, T, 0, stream>>>(x_nodes, emb_strand, emb_pos, xA, N);
  edge_feat_kernel<<<(E * EDGE_DIM + T - 1) / T, T, 0, stream>>>(
      edge_label, edge_sign, emb_strand, emb_sign, We, be, efb, E);

  for (int l = 0; l < LAYERS; ++l) {
    const float* Wql = Wq    + (size_t)l * HID * HID;
    const float* Wkl = Wk    + (size_t)l * HID * HID;
    const float* Wvl = Wv    + (size_t)l * HID * HID;
    const float* Wsl = Wskip + (size_t)l * HID * HID;
    const float* Wel = Wedge + (size_t)l * EDGE_DIM * HID;
    const float* bql = bq    + (size_t)l * HID;
    const float* bkl = bk    + (size_t)l * HID;
    const float* bvl = bv    + (size_t)l * HID;
    const float* bsl = bskip + (size_t)l * HID;

    init_ms_kernel<<<(N * HEADS + T - 1) / T, T, 0, stream>>>(mbuf, sbuf, N * HEADS);
    qkvs_wmma_kernel<<<(N + 15) / 16, 256, 0, stream>>>(
        xA, Wql, bql, Wkl, bkl, Wvl, bvl, Wsl, bsl, qbuf, kbuf, vbuf, xB, N);
    edge_alpha_kernel<<<(E * HEADS + T - 1) / T, T, 0, stream>>>(
        edge_index, qbuf, kbuf, efb, Wel, abuf, mbuf, E);
    edge_expsum_kernel<<<(E * HEADS + T - 1) / T, T, 0, stream>>>(
        edge_index, mbuf, abuf, sbuf, E);
    edge_message_kernel<<<(E * HID + T - 1) / T, T, 0, stream>>>(
        edge_index, vbuf, efb, Wel, abuf, sbuf, xB, E);
    if (l < LAYERS - 1)
      ln_relu_kernel<<<(N + 127) / 128, 128, 0, stream>>>(
          xB, ln_g + (size_t)l * HID, ln_b + (size_t)l * HID, xA, N);
  }

  float* outp = (float*)d_out;
  policy_kernel<<<(N + T - 1) / T, T, 0, stream>>>(xB, Wp, bp, outp, N);
  zero_kernel<<<1, HID, 0, stream>>>(gsum, HID);
  colsum_kernel<<<64, HID, 0, stream>>>(xB, gsum, N);
  value_kernel<<<1, HID, 0, stream>>>(gsum, Wv1, bv1, Wv2, bv2, outp + 2 * (size_t)N, N);
}
